// Encoder_50508815401075
// MI455X (gfx1250) — matrix operand order repaired and checked by reading, not compiled, not run on hardware
//
#include <hip/hip_runtime.h>
#include <hip/hip_bf16.h>

typedef __attribute__((ext_vector_type(16))) __bf16 v16bf;
typedef __attribute__((ext_vector_type(8)))  __bf16 v8bf;
typedef __attribute__((ext_vector_type(8)))  float  v8f;

constexpr int kV  = 32000;   // vocab
constexpr int kE  = 256;     // embedding dim
constexpr int kH  = 1024;    // hidden
constexpr int kB  = 64;      // batch
constexpr int kS  = 512;     // seq len
constexpr int k3H = 3 * kH;  // 3072
constexpr int kNB = 64;      // blocks in persistent scan kernel

// ---------------------------------------------------------------------------
// Prep kernels
// ---------------------------------------------------------------------------
__global__ void k_f32_to_bf16(const float* __restrict__ src,
                              __bf16* __restrict__ dst, int n) {
  int i = blockIdx.x * blockDim.x + threadIdx.x;
  if (i < n) dst[i] = (__bf16)src[i];
}

// Zero h0 (f32 + bf16) and the grid-barrier counter. Deterministic per call.
__global__ void k_init_state(float* __restrict__ hf0, __bf16* __restrict__ hb0,
                             unsigned* __restrict__ cnt) {
  int i = blockIdx.x * blockDim.x + threadIdx.x;
  if (i == 0) *cnt = 0u;
  if (i < kB * kH) {
    hf0[i] = 0.0f;
    hb0[i] = (__bf16)0.0f;
  }
}

// Pack Wx[E,3H] f32 -> bf16 B-fragments, ngroup-major streaming layout:
//   tile (kc, nt) with nt = ng*8 + jt stored at tile index ng*64 + kc*8 + jt.
// Within a tile: lane 0-15 -> col nt*16+lane,  j -> K = kc*32 + {0..7,16..23}[j]
//                lane 16-31 -> col nt*16+lane-16, j -> K = kc*32 + {8..15,24..31}[j]
__global__ void k_pack_wx(const float* __restrict__ W,
                          __bf16* __restrict__ dst) {
  int id = blockIdx.x * blockDim.x + threadIdx.x;
  if (id >= kE * k3H) return;
  int j    = id & 15;
  int lane = (id >> 4) & 31;
  int tmp  = id >> 9;          // ng*64 + kc*8 + jt
  int jt   = tmp & 7;
  int kc   = (tmp >> 3) & 7;
  int ng   = tmp >> 6;
  int nt   = ng * 8 + jt;
  int k    = kc * 32 + ((lane >> 4) << 3) + (j & 7) + ((j >> 3) << 4);
  int col  = nt * 16 + (lane & 15);
  dst[id] = (__bf16)W[k * k3H + col];
}

// Pack Wh[H,3H] f32 -> bf16 B-fragments, column-tile-major layout so each
// scan block's slice is one contiguous 96KB stream (staged into LDS):
//   tile index = (ct*32 + kc)*3 + g ; gate g covers cols g*H + ct*16 ..+16.
__global__ void k_pack_wh(const float* __restrict__ W,
                          __bf16* __restrict__ dst) {
  int id = blockIdx.x * blockDim.x + threadIdx.x;
  if (id >= kH * k3H) return;
  int j    = id & 15;
  int lane = (id >> 4) & 31;
  int tmp  = id >> 9;          // (ct*32 + kc)*3 + g
  int g    = tmp % 3;
  int kk   = tmp / 3;
  int kc   = kk & 31;
  int ct   = kk >> 5;
  int k    = kc * 32 + ((lane >> 4) << 3) + (j & 7) + ((j >> 3) << 4);
  int col  = g * kH + ct * 16 + (lane & 15);
  dst[id] = (__bf16)W[k * k3H + col];
}

// Build A fragment (16x32 bf16, documented CDNA5 layout) from two 16B runs.
__device__ inline v16bf mk_a(v8bf lo, v8bf hi) {
  return __builtin_shufflevector(lo, hi, 0, 1, 2, 3, 4, 5, 6, 7, 8, 9, 10, 11,
                                 12, 13, 14, 15);
}

// ---------------------------------------------------------------------------
// xp = emb[tokens] @ Wx + b0   -> [B*S, 3H] f32
// One wave: one M-tile (16 rows) x 8 N-tiles, K=256 (8 chunks). A prefetched
// one chunk ahead; B addressed with immediate offsets off one base pointer.
// Massively parallel (49152 waves) -> occupancy hides residual load latency.
// ---------------------------------------------------------------------------
__global__ __launch_bounds__(128) void k_xp_gemm(
    const int* __restrict__ tokens, const __bf16* __restrict__ embb,
    const __bf16* __restrict__ WxP, const float* __restrict__ b0,
    float* __restrict__ xp) {
  const int lane  = threadIdx.x & 31;
  const int wave  = threadIdx.x >> 5;
  const int task  = blockIdx.x * 4 + wave;
  const int mtile = task / 24;
  const int ngrp  = task % 24;

  const int row = mtile * 16 + (lane & 15);
  const int tok = tokens[row];
  // lane 0-15: K {k0..k0+7, k0+16..23}; lane 16-31: K {k0+8..15, k0+24..31}
  const __bf16* ap = embb + (size_t)tok * kE + (((lane >> 4) & 1) << 3);
  const __bf16* bbase = WxP + (size_t)ngrp * 64 * 512 + lane * 16;

  v8f acc[8] = {};

  v8bf alo = *(const v8bf*)(ap);
  v8bf ahi = *(const v8bf*)(ap + 16);
#pragma unroll
  for (int kc = 0; kc < 8; ++kc) {
    v16bf a = mk_a(alo, ahi);
    if (kc + 1 < 8) {
      alo = *(const v8bf*)(ap + (kc + 1) * 32);
      ahi = *(const v8bf*)(ap + (kc + 1) * 32 + 16);
    }
#pragma unroll
    for (int j = 0; j < 8; ++j) {
      v16bf bf = *(const v16bf*)(bbase + ((kc * 8 + j) << 9));
      acc[j] = __builtin_amdgcn_wmma_f32_16x16x32_bf16(
          false, a, false, bf, (short)0, acc[j], false, false);
    }
  }

  // C/D layout: lane 0-15 col=lane rows 0..7; lane 16-31 col=lane-16 rows 8..15
  const int rbase = mtile * 16 + (((lane >> 4) & 1) << 3);
#pragma unroll
  for (int j = 0; j < 8; ++j) {
    const int c = (ngrp * 8 + j) * 16 + (lane & 15);
    const float bias = b0[c];
#pragma unroll
    for (int r = 0; r < 8; ++r)
      xp[(size_t)(rbase + r) * k3H + c] = acc[j][r] + bias;
  }
}

// ---------------------------------------------------------------------------
// Persistent GRU scan: 64 blocks x 128 threads run ALL 512 timesteps.
//  - Block ct owns h-columns [ct*16, ct*16+16); its packed Wh slice (96KB)
//    is staged into LDS once and reused every step (ds_load_b128 B operands).
//  - Wave w owns batch rows [16w,16w+16); 3 f32 accumulators (z,r,h gates).
//  - A operand (h state bf16) double-buffered from global (L2 resident).
//  - Steps separated by a monotonic-counter grid barrier; the wait side uses
//    CDNA5 MWAIT (global_load_monitor_b32 + s_monitor_sleep) so the waiting
//    wave sleeps until GL2 sees the counter cacheline written. Falls back to
//    a plain polling spin on hardware without the monitor unit.
// ---------------------------------------------------------------------------
__global__ __launch_bounds__(128) void k_gru_scan(
    const int* __restrict__ tokens, const float* __restrict__ xp,
    const __bf16* __restrict__ WhP, const float* __restrict__ b1,
    float* __restrict__ hf0, __bf16* __restrict__ hb0,
    float* __restrict__ hf1, __bf16* __restrict__ hb1,
    float* __restrict__ seq, float* __restrict__ hT,
    unsigned* __restrict__ bar) {
  extern __shared__ __bf16 lb[];  // 3*32*512 bf16 = 96KB
  const int tid  = threadIdx.x;
  const int lane = tid & 31;
  const int wave = tid >> 5;
  const int ct   = blockIdx.x;

  // Stage this block's Wh slice into LDS (contiguous 96KB copy).
  {
    const uint4* src = (const uint4*)(WhP + (size_t)ct * (3 * 32 * 512));
    uint4* dst = (uint4*)lb;
    for (int i = tid; i < (3 * 32 * 512) / 8; i += 128) dst[i] = src[i];
  }
  __syncthreads();

  const int m    = wave * 16 + (lane & 15);  // batch row for A operand
  const int col  = ct * 16 + (lane & 15);
  const int asel = ((lane >> 4) & 1) << 3;
  const float bz  = b1[col];
  const float brr = b1[kH + col];
  const float bh  = b1[2 * kH + col];
  const int rbase = wave * 16 + (((lane >> 4) & 1) << 3);

  for (int t = 0; t < kS; ++t) {
    const float*  hpf = (t & 1) ? hf1 : hf0;
    const __bf16* hpb = (t & 1) ? hb1 : hb0;
    float*  hnf = (t & 1) ? hf0 : hf1;
    __bf16* hnb = (t & 1) ? hb0 : hb1;
    const bool last = (t == kS - 1);

    const __bf16* ap = hpb + (size_t)m * kH + asel;
    v8f az = {}, ar = {}, ag = {};

    // Software-pipelined K loop: prefetch next A chunk, B from LDS.
    v8bf alo = *(const v8bf*)(ap);
    v8bf ahi = *(const v8bf*)(ap + 16);
#pragma unroll 4
    for (int kc = 0; kc < 32; ++kc) {
      v16bf a = mk_a(alo, ahi);
      if (kc + 1 < 32) {
        alo = *(const v8bf*)(ap + (kc + 1) * 32);
        ahi = *(const v8bf*)(ap + (kc + 1) * 32 + 16);
      }
      const __bf16* lp = lb + kc * 1536 + lane * 16;
      v16bf bzf = *(const v16bf*)(lp);
      v16bf brf = *(const v16bf*)(lp + 512);
      v16bf bgf = *(const v16bf*)(lp + 1024);
      az = __builtin_amdgcn_wmma_f32_16x16x32_bf16(false, a, false, bzf,
                                                   (short)0, az, false, false);
      ar = __builtin_amdgcn_wmma_f32_16x16x32_bf16(false, a, false, brf,
                                                   (short)0, ar, false, false);
      ag = __builtin_amdgcn_wmma_f32_16x16x32_bf16(false, a, false, bgf,
                                                   (short)0, ag, false, false);
    }

    // Gate epilogue + masked state update.
#pragma unroll
    for (int r = 0; r < 8; ++r) {
      const int bi = rbase + r;
      const size_t xbase = ((size_t)bi * kS + t) * k3H;
      const float xz = xp[xbase + col];
      const float xr = xp[xbase + kH + col];
      const float xg = xp[xbase + 2 * kH + col];
      const float hp = hpf[(size_t)bi * kH + col];

      const float z  = 1.0f / (1.0f + __expf(-(xz + az[r] + bz)));
      const float rg = 1.0f / (1.0f + __expf(-(xr + ar[r] + brr)));
      const float hh = tanhf(xg + rg * (ag[r] + bh));
      float hn = z * hp + (1.0f - z) * hh;
      if (tokens[(size_t)bi * kS + t] == 0) hn = hp;  // mask: carry state

      seq[((size_t)bi * kS + t) * kH + col] = hn;
      hnf[(size_t)bi * kH + col] = hn;
      hnb[(size_t)bi * kH + col] = (__bf16)hn;
      if (last) hT[(size_t)bi * kH + col] = hn;
    }

    // Grid barrier between steps (skip after the last one).
    if (!last) {
      __syncthreads();
      if (tid == 0) {
        // Arrive: release-ordered increment (orders the h/seq stores above).
        __hip_atomic_fetch_add(bar, 1u, __ATOMIC_ACQ_REL,
                               __HIP_MEMORY_SCOPE_AGENT);
        const unsigned tgt = (unsigned)kNB * (unsigned)(t + 1);
        // Wait: monitor the counter cacheline in GL2; sleep until written.
        for (;;) {
          unsigned v;
          asm volatile(
              "global_load_monitor_b32 %0, %1, off scope:SCOPE_DEV\n\t"
              "s_wait_loadcnt 0x0"
              : "=v"(v)
              : "v"(bar)
              : "memory");
          if (v >= tgt) break;
          asm volatile("s_monitor_sleep 8" ::: "memory");
        }
        // Acquire: invalidate WGP-level cache before reading other blocks' h.
        __builtin_amdgcn_fence(__ATOMIC_ACQUIRE, "agent");
      }
      __syncthreads();
    }
  }
}

// ---------------------------------------------------------------------------
extern "C" void kernel_launch(void* const* d_in, const int* in_sizes, int n_in,
                              void* d_out, int out_size, void* d_ws,
                              size_t ws_size, hipStream_t stream) {
  const int*   tokens = (const int*)d_in[0];
  const float* emb    = (const float*)d_in[1];
  const float* Wx     = (const float*)d_in[2];
  const float* Wh     = (const float*)d_in[3];
  const float* b      = (const float*)d_in[4];

  char* ws = (char*)d_ws;
  size_t off = 0;
  auto carve = [&](size_t bytes) -> void* {
    void* p = ws + off;
    off = (off + bytes + 255) & ~(size_t)255;
    return p;
  };
  float*    xp   = (float*)carve((size_t)kB * kS * k3H * sizeof(float));
  __bf16*   embb = (__bf16*)carve((size_t)kV * kE * sizeof(__bf16));
  __bf16*   WxP  = (__bf16*)carve((size_t)kE * k3H * sizeof(__bf16));
  __bf16*   WhP  = (__bf16*)carve((size_t)kH * k3H * sizeof(__bf16));
  float*    hf0  = (float*)carve((size_t)kB * kH * sizeof(float));
  float*    hf1  = (float*)carve((size_t)kB * kH * sizeof(float));
  __bf16*   hb0  = (__bf16*)carve((size_t)kB * kH * sizeof(__bf16));
  __bf16*   hb1  = (__bf16*)carve((size_t)kB * kH * sizeof(__bf16));
  unsigned* bar  = (unsigned*)carve(256);

  // Prep: bf16 embedding, packed bf16 weights, zero h0 + barrier counter.
  k_f32_to_bf16<<<(kV * kE + 255) / 256, 256, 0, stream>>>(emb, embb, kV * kE);
  k_pack_wx<<<(kE * k3H + 255) / 256, 256, 0, stream>>>(Wx, WxP);
  k_pack_wh<<<(kH * k3H + 255) / 256, 256, 0, stream>>>(Wh, WhP);
  k_init_state<<<(kB * kH + 255) / 256, 256, 0, stream>>>(hf0, hb0, bar);

  // Big input-projection GEMM.
  k_xp_gemm<<<12288, 128, 0, stream>>>(tokens, embb, WxP, b, xp);

  // Persistent recurrent scan: one launch for all 512 steps.
  float* seq = (float*)d_out;
  float* hT  = seq + (size_t)kB * kS * kH;
  k_gru_scan<<<kNB, 128, 3 * 32 * 512 * sizeof(__bf16), stream>>>(
      tokens, xp, WhP, b + k3H, hf0, hb0, hf1, hb1, seq, hT, bar);
}
